// CrossTaskKnowledgeDistiller_87677462380784
// MI455X (gfx1250) — matrix operand order repaired and checked by reading, not compiled, not run on hardware
//
#include <hip/hip_runtime.h>

typedef __attribute__((ext_vector_type(16))) _Float16 v16h;
typedef __attribute__((ext_vector_type(2)))  _Float16 v2h;
typedef __attribute__((ext_vector_type(8)))  float    v8f;
typedef __attribute__((ext_vector_type(4)))  int      v4i;

#define BATCH 8
#define CH    128
#define HW    3136          // 56*56 = 196 tiles of 16 = 98 chunks of 32

#define GAS __attribute__((address_space(1)))

// force a global (address_space 1) 16-byte load -> global_load_b128
__device__ __forceinline__ v4i gload_b128(const void* p) {
    return *(const GAS v4i*)(unsigned long long)p;
}
__device__ __forceinline__ float gload_f32(const float* p) {
    return *(const GAS float*)(unsigned long long)p;
}

// ---------------- Stage 1: 1x1-conv projections (minor cost, VALU) ----------------

__global__ void transpose128_k(const float* __restrict__ W, float* __restrict__ Wt) {
    int i = blockIdx.x * 256 + threadIdx.x;       // 16384 total
    int r = i >> 7, c = i & 127;
    Wt[c * 128 + r] = W[r * 128 + c];
}

// out[b][hw][o] = f16( (sum_c X[b][c][hw]*Wt[c][o] + bias[o]) * premul )
__global__ void proj_row_k(const float* __restrict__ X, const float* __restrict__ Wt,
                           const float* __restrict__ bias, _Float16* __restrict__ out,
                           float premul) {
    int idx = blockIdx.x * 256 + threadIdx.x;     // B*HW*C, o fastest
    int o   = idx & 127;
    int rest = idx >> 7;
    int hw  = rest % HW;
    int b   = rest / HW;
    const float* x = X + (size_t)b * CH * HW + hw;
    float acc = bias[o];
#pragma unroll 8
    for (int c = 0; c < CH; ++c)
        acc += gload_f32(x + (size_t)c * HW) * gload_f32(Wt + c * 128 + o);
    out[idx] = (_Float16)(acc * premul);
}

// outT[b][o][hw] = f16( sum_c X[b][c][hw]*W[o][c] + bias[o] )   (transposed store)
__global__ void proj_vT_k(const float* __restrict__ X, const float* __restrict__ W,
                          const float* __restrict__ bias, _Float16* __restrict__ outT) {
    int idx = blockIdx.x * 256 + threadIdx.x;     // B*C*HW, hw fastest
    int hw  = idx % HW;
    int rest = idx / HW;
    int o   = rest & 127;
    int b   = rest >> 7;
    const float* x = X + (size_t)b * CH * HW + hw;
    float acc = bias[o];
#pragma unroll 8
    for (int c = 0; c < CH; ++c)
        acc += gload_f32(x + (size_t)c * HW) * gload_f32(W + o * 128 + c);
    outT[idx] = (_Float16)acc;
}

// ---------------- Stage 2: flash attention on WMMA ----------------

__device__ __forceinline__ v8f wmma16(v16h a, v16h b, v8f c) {
    return __builtin_amdgcn_wmma_f32_16x16x32_f16(false, a, false, b, (short)0, c,
                                                  false, false);
}

// A fragment (16x32 f16, M=row per lane, K half-interleave granularity 16):
//   lane(m,h): VGPR j -> K = coff+16h+2j,+1  => 32 contiguous bytes
__device__ __forceinline__ v16h load_a_frag(const _Float16* rowptr, int coff, int h) {
    union { v16h v; v4i q[2]; } u;
    const char* p = (const char*)(rowptr + coff) + 32 * h;
    u.q[0] = gload_b128(p);
    u.q[1] = gload_b128(p + 16);
    return u.v;
}

// B fragment (32x16 f16, N=col per lane, K half-interleave granularity 16):
//   lane(n,h): VGPR j -> K = coff+16h+2j,+1  => 32 contiguous bytes
__device__ __forceinline__ v16h load_b_frag(const _Float16* rowptr, int coff, int h) {
    union { v16h v; v4i q[2]; } u;
    const char* p = (const char*)(rowptr + coff) + 32 * h;
    u.q[0] = gload_b128(p);
    u.q[1] = gload_b128(p + 16);
    return u.v;
}

__device__ __forceinline__ int pack2_f16(float a, float b) {
    union { v2h h; int i; } u;
    u.h = v2h{(_Float16)a, (_Float16)b};
    return u.i;
}

__global__ __launch_bounds__(128)
void flash_attn_k(const _Float16* __restrict__ Qh,
                  const _Float16* __restrict__ Kh1, const _Float16* __restrict__ Vt1,
                  const _Float16* __restrict__ Kh2, const _Float16* __restrict__ Vt2,
                  const float* __restrict__ s, float* __restrict__ out) {
    const int lane  = threadIdx.x & 31;
    const int n     = lane & 15;        // query (col) index inside tile / tile row
    const int h     = lane >> 4;        // wave half
    const int gwave = blockIdx.x * 4 + (threadIdx.x >> 5);
    const int b     = gwave / (HW / 16);
    const int q0    = (gwave % (HW / 16)) * 16;

    const _Float16* Qb = Qh + (size_t)b * HW * CH;
    const _Float16* Kt[2] = { Kh1 + (size_t)b * HW * CH, Kh2 + (size_t)b * HW * CH };
    const _Float16* Vt[2] = { Vt1 + (size_t)b * CH * HW, Vt2 + (size_t)b * CH * HW };

    const GAS float* sg = (const GAS float*)(unsigned long long)s;
    GAS float*       og = (GAS float*)(unsigned long long)out;

    // Preload this lane's query row as B fragments for S^T = K * Q^T
    v16h qf[4];
#pragma unroll
    for (int cc = 0; cc < 4; ++cc)
        qf[cc] = load_b_frag(Qb + (size_t)(q0 + n) * CH, 32 * cc, h);

#pragma unroll 1
    for (int t = 0; t < 2; ++t) {
        const _Float16* K = Kt[t];
        const _Float16* V = Vt[t];

        v8f O[8];
#pragma unroll
        for (int i = 0; i < 8; ++i) O[i] = v8f{};
        float m = -1e30f, l = 0.f;

#pragma unroll 1
        for (int kc = 0; kc < HW; kc += 32) {
            // ---- S^T (32 keys x 16 queries), accumulated over C in 4 chunks ----
            v8f st0 = v8f{}, st1 = v8f{};
#pragma unroll
            for (int cc = 0; cc < 4; ++cc) {
                v16h a0 = load_a_frag(K + (size_t)(kc + n) * CH,      32 * cc, h);
                v16h a1 = load_a_frag(K + (size_t)(kc + 16 + n) * CH, 32 * cc, h);
                st0 = wmma16(a0, qf[cc], st0);
                st1 = wmma16(a1, qf[cc], st1);
            }
            // ---- online softmax (per-lane == per-query; log2 domain) ----
            float mx = st0[0];
#pragma unroll
            for (int r = 1; r < 8; ++r) mx = fmaxf(mx, st0[r]);
#pragma unroll
            for (int r = 0; r < 8; ++r) mx = fmaxf(mx, st1[r]);
            mx = fmaxf(mx, __shfl_xor(mx, 16, 32));
            float mnew  = fmaxf(m, mx);
            float alpha = __builtin_amdgcn_exp2f(m - mnew);
            float rs = 0.f;
#pragma unroll
            for (int r = 0; r < 8; ++r) {
                st0[r] = __builtin_amdgcn_exp2f(st0[r] - mnew); rs += st0[r];
                st1[r] = __builtin_amdgcn_exp2f(st1[r] - mnew); rs += st1[r];
            }
            rs += __shfl_xor(rs, 16, 32);
            l = l * alpha + rs;
            m = mnew;
#pragma unroll
            for (int i = 0; i < 8; ++i)
#pragma unroll
                for (int r = 0; r < 8; ++r) O[i][r] *= alpha;

            // ---- repack P^T (D layout) into B-fragment layout ----
            int pk0[4], pk1[4], o0[4], o1[4];
#pragma unroll
            for (int j = 0; j < 4; ++j) {
                pk0[j] = pack2_f16(st0[2 * j], st0[2 * j + 1]);  // k = 2j+8h'
                pk1[j] = pack2_f16(st1[2 * j], st1[2 * j + 1]);  // k = 16+2j+8h'
            }
#pragma unroll
            for (int j = 0; j < 4; ++j) {
                o0[j] = __shfl_xor(pk0[j], 16, 32);
                o1[j] = __shfl_xor(pk1[j], 16, 32);
            }
            union { v16h v; int i[8]; } bf;
#pragma unroll
            for (int j = 0; j < 4; ++j) {
                bf.i[j]     = h ? o1[j]  : pk0[j];   // K = 2j   + 16h
                bf.i[4 + j] = h ? pk1[j] : o0[j];    // K = 8+2j + 16h
            }
            // ---- O^T += V^T * P^T  (8 channel tiles) ----
#pragma unroll
            for (int ct = 0; ct < 8; ++ct) {
                v16h va = load_a_frag(V + (size_t)(ct * 16 + n) * HW, kc, h);
                O[ct] = wmma16(va, bf.v, O[ct]);
            }
        }

        // ---- epilogue: out = s + 0.5 * (O1/l1 + O2/l2) ----
        float invl = 0.5f / l;
#pragma unroll
        for (int ct = 0; ct < 8; ++ct)
#pragma unroll
            for (int r = 0; r < 8; ++r) {
                int c = ct * 16 + r + 8 * h;
                size_t idx = ((size_t)b * CH + c) * HW + q0 + n;
                float val = O[ct][r] * invl;
                og[idx] = (t == 0) ? (sg[idx] + val) : (og[idx] + val);
            }
    }
}

// ---------------- host launcher ----------------

extern "C" void kernel_launch(void* const* d_in, const int* in_sizes, int n_in,
                              void* d_out, int out_size, void* d_ws, size_t ws_size,
                              hipStream_t stream) {
    const float* s  = (const float*)d_in[0];
    const float* t1 = (const float*)d_in[1];
    const float* t2 = (const float*)d_in[2];
    const float* wq = (const float*)d_in[3];
    const float* bq = (const float*)d_in[4];
    const float* wk = (const float*)d_in[5];
    const float* bk = (const float*)d_in[6];
    const float* wv = (const float*)d_in[7];
    const float* bv = (const float*)d_in[8];
    float* out = (float*)d_out;

    const size_t SZ = (size_t)BATCH * HW * CH;          // 3,211,264 elements
    _Float16* Qh  = (_Float16*)d_ws;
    _Float16* Kh1 = Qh  + SZ;
    _Float16* Kh2 = Kh1 + SZ;
    _Float16* Vt1 = Kh2 + SZ;
    _Float16* Vt2 = Vt1 + SZ;
    float*    wq_t = (float*)(Vt2 + SZ);
    float*    wk_t = wq_t + 128 * 128;

    // fold softmax scale (C^-1/2) and log2(e) into Q so exp() becomes exp2()
    const float qpremul = 0.08838834764831845f * 1.44269504088896340f;

    transpose128_k<<<64, 256, 0, stream>>>(wq, wq_t);
    transpose128_k<<<64, 256, 0, stream>>>(wk, wk_t);

    const int PG = (int)(SZ / 256);                     // 12544 blocks
    proj_row_k<<<PG, 256, 0, stream>>>(s,  wq_t, bq, Qh,  qpremul);
    proj_row_k<<<PG, 256, 0, stream>>>(t1, wk_t, bk, Kh1, 1.0f);
    proj_row_k<<<PG, 256, 0, stream>>>(t2, wk_t, bk, Kh2, 1.0f);
    proj_vT_k<<<PG, 256, 0, stream>>>(t1, wv, bv, Vt1);
    proj_vT_k<<<PG, 256, 0, stream>>>(t2, wv, bv, Vt2);

    // 8 batches * 196 query tiles = 1568 waves = 392 blocks of 4 waves
    flash_attn_k<<<392, 128, 0, stream>>>(Qh, Kh1, Vt1, Kh2, Vt2, s, out);
}
